// GCN_GIN_53498112639140
// MI455X (gfx1250) — matrix operand order repaired and checked by reading, not compiled, not run on hardware
//
#include <hip/hip_runtime.h>
#include <hip/hip_bf16.h>
#include <stdint.h>

typedef __bf16 bf16;
typedef __attribute__((ext_vector_type(16))) __bf16        v16bf;
typedef __attribute__((ext_vector_type(8)))  float         v8f;
typedef __attribute__((ext_vector_type(8)))  unsigned int  v8u;

#define TPB 256

// ---------------------------------------------------------------------------
// Elementwise / setup kernels
// ---------------------------------------------------------------------------
__global__ void k_fill_zero(float* __restrict__ p, int n) {
  int i = blockIdx.x * blockDim.x + threadIdx.x;
  if (i < n) p[i] = 0.0f;
}

__global__ void k_degree(const int* __restrict__ dst, float* __restrict__ deg, int E) {
  int e = blockIdx.x * blockDim.x + threadIdx.x;
  if (e < E) atomicAdd(&deg[dst[e]], 1.0f);
}

__global__ void k_dinv(const float* __restrict__ deg, float* __restrict__ dinv, int n) {
  int i = blockIdx.x * blockDim.x + threadIdx.x;
  if (i < n) dinv[i] = rsqrtf(deg[i] + 1.0f);
}

__global__ void k_f32_to_bf16_pad(const float* __restrict__ in, bf16* __restrict__ out,
                                  int n, int F, int Fpad) {
  int i = blockIdx.x * blockDim.x + threadIdx.x;
  if (i >= n * Fpad) return;
  int r = i / Fpad, f = i - r * Fpad;
  out[i] = (f < F) ? (bf16)in[(size_t)r * F + f] : (bf16)0.0f;
}

__global__ void k_bf16_to_f32(const bf16* __restrict__ in, float* __restrict__ out, int n) {
  int i = blockIdx.x * blockDim.x + threadIdx.x;
  if (i < n) out[i] = (float)in[i];
}

__global__ void k_copy_f32(const float* __restrict__ in, float* __restrict__ out, int n) {
  int i = blockIdx.x * blockDim.x + threadIdx.x;
  if (i < n) out[i] = in[i];
}

// W [K x Nout] f32 (row major)  ->  Wt [Nout x Kpad] bf16 (transposed, K padded)
__global__ void k_wT_bf16(const float* __restrict__ W, bf16* __restrict__ Wt,
                          int K, int Kpad, int Nout) {
  int i = blockIdx.x * blockDim.x + threadIdx.x;
  if (i >= Nout * Kpad) return;
  int nn = i / Kpad, k = i - nn * Kpad;
  Wt[i] = (k < K) ? (bf16)W[(size_t)k * Nout + nn] : (bf16)0.0f;
}

// ---------------------------------------------------------------------------
// WMMA GEMM:  C[M x Nout] = A[M x KPAD](bf16) @ Wt^T   (+bias, optional relu)
// One wave computes a 16x64 output tile (A fragment reused across 4 WMMAs).
// KPAD is a compile-time constant so the K loop fully unrolls: the scheduler
// can hoist each K-step's 10 b128 loads ahead of earlier WMMAs with distinct
// registers (software pipelining without rotation movs).
// Requires M % 16 == 0, KPAD % 32 == 0, Nout % 64 == 0.
// Fragment layouts per CDNA5 ISA 7.12.2 (16-bit A 16x32 / B 32x16 / f32 C 16x16).
// ---------------------------------------------------------------------------
template <int KPAD>
__global__ void k_wmma_gemm(const bf16* __restrict__ A, const bf16* __restrict__ Bt,
                            const float* __restrict__ bias, float* __restrict__ C,
                            int M, int Nout, int relu) {
  int gtid  = blockIdx.x * blockDim.x + threadIdx.x;
  int wave  = gtid >> 5;
  int lane  = threadIdx.x & 31;
  int ntiles = Nout >> 6;            // tiles of 64 columns
  int mtiles = M >> 4;
  if (wave >= mtiles * ntiles) return;   // wave-uniform: EXEC stays all-ones
  int mtile = wave / ntiles;
  int ntile = wave - mtile * ntiles;
  int lm    = lane & 15;
  int khalf = lane >> 4;

  const bf16* Arow = A + (size_t)(mtile * 16 + lm) * KPAD;

  v8f acc[4] = {};
#pragma unroll
  for (int kk = 0; kk < KPAD; kk += 32) {
    // A fragment: lane holds row m=lm; K = kk + 16*khalf + 0..15 (contiguous)
    const uint4* ap0 = (const uint4*)(Arow + kk + 16 * khalf);
    const uint4* ap1 = (const uint4*)(Arow + kk + 16 * khalf + 8);
    uint4 a0 = ap0[0];
    uint4 a1 = ap1[0];
    v8u au;
    au[0] = a0.x; au[1] = a0.y; au[2] = a0.z; au[3] = a0.w;
    au[4] = a1.x; au[5] = a1.y; au[6] = a1.z; au[7] = a1.w;
    v16bf af = __builtin_bit_cast(v16bf, au);

#pragma unroll
    for (int t = 0; t < 4; ++t) {
      // B fragment: lane holds col n; K = kk + 16*khalf + 0..15 (contiguous in Wt row)
      int n = (ntile << 6) + (t << 4) + lm;
      const uint4* bp = (const uint4*)(Bt + (size_t)n * KPAD + kk + 16 * khalf);
      uint4 b0 = bp[0];
      uint4 b1 = bp[1];
      v8u bu;
      bu[0] = b0.x; bu[1] = b0.y; bu[2] = b0.z; bu[3] = b0.w;
      bu[4] = b1.x; bu[5] = b1.y; bu[6] = b1.z; bu[7] = b1.w;
      v16bf bfr = __builtin_bit_cast(v16bf, bu);

      acc[t] = __builtin_amdgcn_wmma_f32_16x16x32_bf16(
          /*neg_a=*/false, af, /*neg_b=*/false, bfr,
          /*c_mod=*/(short)0, acc[t], /*reuse_a=*/false, /*reuse_b=*/false);
    }
  }

  // C layout: VGPR r holds row m = r + 8*khalf, col n = lm  (M % 16 == 0)
#pragma unroll
  for (int t = 0; t < 4; ++t) {
    int n = (ntile << 6) + (t << 4) + lm;
    float bv = bias ? bias[n] : 0.0f;
#pragma unroll
    for (int r = 0; r < 8; ++r) {
      int mr = (mtile << 4) + r + 8 * khalf;
      float v = acc[t][r] + bv;
      if (relu) v = fmaxf(v, 0.0f);
      C[(size_t)mr * Nout + n] = v;
    }
  }
}

// ---------------------------------------------------------------------------
// GCN aggregation: agg = D^-1/2 (A+I) D^-1/2 H   (self term + edge scatter)
// ---------------------------------------------------------------------------
__global__ void k_gcn_self(const float* __restrict__ H, const float* __restrict__ dinv,
                           float* __restrict__ agg, int n, int F) {
  int i = blockIdx.x * blockDim.x + threadIdx.x;
  if (i >= n * F) return;
  int r = i / F;
  float d = dinv[r];
  agg[i] = H[i] * d * d;
}

__global__ void k_gcn_scatter(const float* __restrict__ H, const int* __restrict__ src,
                              const int* __restrict__ dst, const float* __restrict__ dinv,
                              float* __restrict__ agg, int E, int F) {
  int F4  = F >> 2;
  int tid = blockIdx.x * blockDim.x + threadIdx.x;
  if (tid >= E * F4) return;
  int e = tid / F4, q = tid - e * F4;
  int s = src[e], d = dst[e];
  float coef = dinv[s] * dinv[d];
  const float4 h = ((const float4*)H)[(size_t)s * F4 + q];
  float* a = agg + (size_t)d * F + q * 4;
  atomicAdd(a + 0, h.x * coef);
  atomicAdd(a + 1, h.y * coef);
  atomicAdd(a + 2, h.z * coef);
  atomicAdd(a + 3, h.w * coef);
}

// +bias, relu, batchnorm(eval)  ->  bf16 activations for next layer (F == Fpad)
__global__ void k_gcn_post(const float* __restrict__ agg, const float* __restrict__ bias,
                           const float* __restrict__ gamma, const float* __restrict__ beta,
                           const float* __restrict__ mean, const float* __restrict__ var,
                           bf16* __restrict__ out, int n, int F) {
  int i = blockIdx.x * blockDim.x + threadIdx.x;
  if (i >= n * F) return;
  int f = i % F;
  float v = agg[i] + bias[f];
  v = fmaxf(v, 0.0f);
  v = (v - mean[f]) * (gamma[f] * rsqrtf(var[f] + 1e-5f)) + beta[f];
  out[i] = (bf16)v;
}

// ---------------------------------------------------------------------------
// GIN aggregation: agg = x + sum_{src->dst} x[src]
// ---------------------------------------------------------------------------
__global__ void k_gin_scatter_f32(const float* __restrict__ X, const int* __restrict__ src,
                                  const int* __restrict__ dst, float* __restrict__ agg,
                                  int E, int F) {
  int tid = blockIdx.x * blockDim.x + threadIdx.x;
  if (tid >= E * F) return;
  int e = tid / F, f = tid - e * F;
  atomicAdd(&agg[(size_t)dst[e] * F + f], X[(size_t)src[e] * F + f]);
}

__global__ void k_gin_scatter_bf16(const bf16* __restrict__ Z, const int* __restrict__ src,
                                   const int* __restrict__ dst, float* __restrict__ agg,
                                   int E, int F) {
  int tid = blockIdx.x * blockDim.x + threadIdx.x;
  if (tid >= E * F) return;
  int e = tid / F, f = tid - e * F;
  atomicAdd(&agg[(size_t)dst[e] * F + f], (float)Z[(size_t)src[e] * F + f]);
}

// relu then batchnorm(eval) -> bf16
__global__ void k_relu_bn_bf16(const float* __restrict__ u,
                               const float* __restrict__ gamma, const float* __restrict__ beta,
                               const float* __restrict__ mean, const float* __restrict__ var,
                               bf16* __restrict__ out, int n, int F) {
  int i = blockIdx.x * blockDim.x + threadIdx.x;
  if (i >= n * F) return;
  int f = i % F;
  float v = fmaxf(u[i], 0.0f);
  v = (v - mean[f]) * (gamma[f] * rsqrtf(var[f] + 1e-5f)) + beta[f];
  out[i] = (bf16)v;
}

// global_add_pool over batch id
__global__ void k_pool(const bf16* __restrict__ Z, const int* __restrict__ batch,
                       float* __restrict__ P, int n, int F) {
  int i = blockIdx.x * blockDim.x + threadIdx.x;
  if (i >= n * F) return;
  int r = i / F, f = i - r * F;
  atomicAdd(&P[(size_t)batch[r] * F + f], (float)Z[i]);
}

// ---------------------------------------------------------------------------
// Tiny head MLP on [64 x 192] -> [64]; one workgroup, LDS staged.
// ---------------------------------------------------------------------------
__global__ void k_mlp(const float* __restrict__ hg, const float* __restrict__ zg,
                      const float* __restrict__ w1, const float* __restrict__ b1,
                      const float* __restrict__ w2, const float* __restrict__ b2,
                      const float* __restrict__ w3, const float* __restrict__ b3,
                      float* __restrict__ out) {
  __shared__ float h1[64 * 64];
  __shared__ float h2[64 * 64];
  int tid = threadIdx.x;
  for (int idx = tid; idx < 64 * 64; idx += blockDim.x) {
    int g = idx >> 6, o = idx & 63;
    float s = b1[o];
    for (int k = 0; k < 128; ++k) s += hg[g * 128 + k] * w1[k * 64 + o];
    for (int k = 0; k < 64; ++k)  s += zg[g * 64 + k] * w1[(128 + k) * 64 + o];
    h1[idx] = fmaxf(s, 0.0f);
  }
  __syncthreads();
  for (int idx = tid; idx < 64 * 64; idx += blockDim.x) {
    int g = idx >> 6, o = idx & 63;
    float s = b2[o];
    for (int k = 0; k < 64; ++k) s += h1[g * 64 + k] * w2[k * 64 + o];
    h2[idx] = fmaxf(s, 0.0f);
  }
  __syncthreads();
  if (tid < 64) {
    float s = b3[0];
    for (int k = 0; k < 64; ++k) s += h2[tid * 64 + k] * w3[k];
    out[tid] = fmaxf(s, 0.0f);
  }
}

// ---------------------------------------------------------------------------
// Host-side orchestration
// ---------------------------------------------------------------------------
static inline dim3 g1(long n) { return dim3((unsigned)((n + TPB - 1) / TPB)); }

extern "C" void kernel_launch(void* const* d_in, const int* in_sizes, int n_in,
                              void* d_out, int out_size, void* d_ws, size_t ws_size,
                              hipStream_t stream) {
  const int F_IN = 373, F_PAD = 384, G = 64;
  const float* x   = (const float*)d_in[0];
  const int* ei    = (const int*)d_in[1];
  const int* batch = (const int*)d_in[2];
  const int N = in_sizes[0] / F_IN;
  const int E = in_sizes[1] / 2;
  const int* src = ei;
  const int* dst = ei + E;

  // params (flattened in dict insertion order)
  const float* conv1w = (const float*)d_in[3];  const float* conv1b = (const float*)d_in[4];
  const float* bn01g = (const float*)d_in[5];   const float* bn01b = (const float*)d_in[6];
  const float* bn01m = (const float*)d_in[7];   const float* bn01v = (const float*)d_in[8];
  const float* conv2w = (const float*)d_in[9];  const float* conv2b = (const float*)d_in[10];
  const float* bn02g = (const float*)d_in[11];  const float* bn02b = (const float*)d_in[12];
  const float* bn02m = (const float*)d_in[13];  const float* bn02v = (const float*)d_in[14];
  const float* conv3w = (const float*)d_in[15]; const float* conv3b = (const float*)d_in[16];
  const float* bn03g = (const float*)d_in[17];  const float* bn03b = (const float*)d_in[18];
  const float* bn03m = (const float*)d_in[19];  const float* bn03v = (const float*)d_in[20];
  const float* g1aw = (const float*)d_in[21];   const float* g1ab = (const float*)d_in[22];
  const float* g1bw = (const float*)d_in[23];   const float* g1bb = (const float*)d_in[24];
  const float* bn21g = (const float*)d_in[25];  const float* bn21b = (const float*)d_in[26];
  const float* bn21m = (const float*)d_in[27];  const float* bn21v = (const float*)d_in[28];
  const float* g2aw = (const float*)d_in[29];   const float* g2ab = (const float*)d_in[30];
  const float* g2bw = (const float*)d_in[31];   const float* g2bb = (const float*)d_in[32];
  const float* bn22g = (const float*)d_in[33];  const float* bn22b = (const float*)d_in[34];
  const float* bn22m = (const float*)d_in[35];  const float* bn22v = (const float*)d_in[36];
  const float* g3aw = (const float*)d_in[37];   const float* g3ab = (const float*)d_in[38];
  const float* g3bw = (const float*)d_in[39];   const float* g3bb = (const float*)d_in[40];
  const float* bn23g = (const float*)d_in[41];  const float* bn23b = (const float*)d_in[42];
  const float* bn23m = (const float*)d_in[43];  const float* bn23v = (const float*)d_in[44];
  const float* fc1w = (const float*)d_in[45];   const float* fc1b = (const float*)d_in[46];
  const float* fc2w = (const float*)d_in[47];   const float* fc2b = (const float*)d_in[48];
  const float* fc3w = (const float*)d_in[49];   const float* fc3b = (const float*)d_in[50];

  // ---- workspace carve ----
  size_t off = 0;
  auto carve = [&](size_t bytes) -> void* {
    void* r = (char*)d_ws + off;
    off += (bytes + 255) & ~(size_t)255;
    return r;
  };
  float* deg   = (float*)carve((size_t)N * 4);
  float* dinv  = (float*)carve((size_t)N * 4);
  float* bufA  = (float*)carve((size_t)N * F_PAD * 4);
  float* bufB  = (float*)carve((size_t)N * F_PAD * 4);
  bf16*  act1  = (bf16*)carve((size_t)N * F_PAD * 2);
  bf16*  act2  = (bf16*)carve((size_t)N * F_PAD * 2);
  bf16* wt_c1  = (bf16*)carve((size_t)256 * 384 * 2);
  bf16* wt_c2  = (bf16*)carve((size_t)128 * 256 * 2);
  bf16* wt_c3  = (bf16*)carve((size_t)128 * 128 * 2);
  bf16* wt_g1a = (bf16*)carve((size_t)256 * 384 * 2);
  bf16* wt_g1b = (bf16*)carve((size_t)256 * 256 * 2);
  bf16* wt_g2a = (bf16*)carve((size_t)128 * 256 * 2);
  bf16* wt_g2b = (bf16*)carve((size_t)128 * 128 * 2);
  bf16* wt_g3a = (bf16*)carve((size_t)64 * 128 * 2);
  bf16* wt_g3b = (bf16*)carve((size_t)64 * 64 * 2);
  float* hg    = (float*)carve((size_t)G * 128 * 4);
  float* zg    = (float*)carve((size_t)G * 64 * 4);

  auto gemm = [&](const bf16* A, const bf16* Wt, const float* bias, float* C,
                  int M, int Kpad, int Nout, int relu) {
    long waves = (long)(M / 16) * (Nout / 64);
    dim3 grid = g1(waves * 32);
    switch (Kpad) {
      case 384: k_wmma_gemm<384><<<grid, TPB, 0, stream>>>(A, Wt, bias, C, M, Nout, relu); break;
      case 256: k_wmma_gemm<256><<<grid, TPB, 0, stream>>>(A, Wt, bias, C, M, Nout, relu); break;
      case 128: k_wmma_gemm<128><<<grid, TPB, 0, stream>>>(A, Wt, bias, C, M, Nout, relu); break;
      default:  k_wmma_gemm<64><<<grid, TPB, 0, stream>>>(A, Wt, bias, C, M, Nout, relu); break;
    }
  };

  // ---- degrees ----
  k_fill_zero<<<g1(N), TPB, 0, stream>>>(deg, N);
  k_degree<<<g1(E), TPB, 0, stream>>>(dst, deg, E);
  k_dinv<<<g1(N), TPB, 0, stream>>>(deg, dinv, N);

  // ---- weight conversion (f32 -> bf16, transposed, K padded to x32) ----
  k_wT_bf16<<<g1(256 * 384), TPB, 0, stream>>>(conv1w, wt_c1, 373, 384, 256);
  k_wT_bf16<<<g1(128 * 256), TPB, 0, stream>>>(conv2w, wt_c2, 256, 256, 128);
  k_wT_bf16<<<g1(128 * 128), TPB, 0, stream>>>(conv3w, wt_c3, 128, 128, 128);
  k_wT_bf16<<<g1(256 * 384), TPB, 0, stream>>>(g1aw, wt_g1a, 373, 384, 256);
  k_wT_bf16<<<g1(256 * 256), TPB, 0, stream>>>(g1bw, wt_g1b, 256, 256, 256);
  k_wT_bf16<<<g1(128 * 256), TPB, 0, stream>>>(g2aw, wt_g2a, 256, 256, 128);
  k_wT_bf16<<<g1(128 * 128), TPB, 0, stream>>>(g2bw, wt_g2b, 128, 128, 128);
  k_wT_bf16<<<g1(64 * 128), TPB, 0, stream>>>(g3aw, wt_g3a, 128, 128, 64);
  k_wT_bf16<<<g1(64 * 64), TPB, 0, stream>>>(g3bw, wt_g3b, 64, 64, 64);

  // ================= GCN branch =================
  k_f32_to_bf16_pad<<<g1((long)N * 384), TPB, 0, stream>>>(x, act1, N, 373, 384);
  // conv1
  gemm(act1, wt_c1, nullptr, bufB, N, 384, 256, 0);
  k_gcn_self<<<g1((long)N * 256), TPB, 0, stream>>>(bufB, dinv, bufA, N, 256);
  k_gcn_scatter<<<g1((long)E * 64), TPB, 0, stream>>>(bufB, src, dst, dinv, bufA, E, 256);
  k_gcn_post<<<g1((long)N * 256), TPB, 0, stream>>>(bufA, conv1b, bn01g, bn01b, bn01m, bn01v,
                                                    act2, N, 256);
  // conv2
  gemm(act2, wt_c2, nullptr, bufB, N, 256, 128, 0);
  k_gcn_self<<<g1((long)N * 128), TPB, 0, stream>>>(bufB, dinv, bufA, N, 128);
  k_gcn_scatter<<<g1((long)E * 32), TPB, 0, stream>>>(bufB, src, dst, dinv, bufA, E, 128);
  k_gcn_post<<<g1((long)N * 128), TPB, 0, stream>>>(bufA, conv2b, bn02g, bn02b, bn02m, bn02v,
                                                    act1, N, 128);
  // conv3
  gemm(act1, wt_c3, nullptr, bufB, N, 128, 128, 0);
  k_gcn_self<<<g1((long)N * 128), TPB, 0, stream>>>(bufB, dinv, bufA, N, 128);
  k_gcn_scatter<<<g1((long)E * 32), TPB, 0, stream>>>(bufB, src, dst, dinv, bufA, E, 128);
  k_gcn_post<<<g1((long)N * 128), TPB, 0, stream>>>(bufA, conv3b, bn03g, bn03b, bn03m, bn03v,
                                                    act2, N, 128);
  // pool -> hg [64 x 128]
  k_fill_zero<<<g1(G * 128), TPB, 0, stream>>>(hg, G * 128);
  k_pool<<<g1((long)N * 128), TPB, 0, stream>>>(act2, batch, hg, N, 128);

  // ================= GIN branch =================
  // gin1: agg = x + scatter(x)
  k_copy_f32<<<g1((long)N * 373), TPB, 0, stream>>>(x, bufA, N * 373);
  k_gin_scatter_f32<<<g1((long)E * 373), TPB, 0, stream>>>(x, src, dst, bufA, E, 373);
  k_f32_to_bf16_pad<<<g1((long)N * 384), TPB, 0, stream>>>(bufA, act1, N, 373, 384);
  gemm(act1, wt_g1a, g1ab, bufB, N, 384, 256, 1);                 // relu fused
  k_f32_to_bf16_pad<<<g1((long)N * 256), TPB, 0, stream>>>(bufB, act2, N, 256, 256);
  gemm(act2, wt_g1b, g1bb, bufA, N, 256, 256, 0);
  k_relu_bn_bf16<<<g1((long)N * 256), TPB, 0, stream>>>(bufA, bn21g, bn21b, bn21m, bn21v,
                                                        act1, N, 256);
  // gin2
  k_bf16_to_f32<<<g1((long)N * 256), TPB, 0, stream>>>(act1, bufB, N * 256);
  k_gin_scatter_bf16<<<g1((long)E * 256), TPB, 0, stream>>>(act1, src, dst, bufB, E, 256);
  k_f32_to_bf16_pad<<<g1((long)N * 256), TPB, 0, stream>>>(bufB, act2, N, 256, 256);
  gemm(act2, wt_g2a, g2ab, bufA, N, 256, 128, 1);
  k_f32_to_bf16_pad<<<g1((long)N * 128), TPB, 0, stream>>>(bufA, act1, N, 128, 128);
  gemm(act1, wt_g2b, g2bb, bufB, N, 128, 128, 0);
  k_relu_bn_bf16<<<g1((long)N * 128), TPB, 0, stream>>>(bufB, bn22g, bn22b, bn22m, bn22v,
                                                        act2, N, 128);
  // gin3
  k_bf16_to_f32<<<g1((long)N * 128), TPB, 0, stream>>>(act2, bufA, N * 128);
  k_gin_scatter_bf16<<<g1((long)E * 128), TPB, 0, stream>>>(act2, src, dst, bufA, E, 128);
  k_f32_to_bf16_pad<<<g1((long)N * 128), TPB, 0, stream>>>(bufA, act1, N, 128, 128);
  gemm(act1, wt_g3a, g3ab, bufB, N, 128, 64, 1);
  k_f32_to_bf16_pad<<<g1((long)N * 64), TPB, 0, stream>>>(bufB, act2, N, 64, 64);
  gemm(act2, wt_g3b, g3bb, bufA, N, 64, 64, 0);
  k_relu_bn_bf16<<<g1((long)N * 64), TPB, 0, stream>>>(bufA, bn23g, bn23b, bn23m, bn23v,
                                                       act1, N, 64);
  // pool -> zg [64 x 64]
  k_fill_zero<<<g1(G * 64), TPB, 0, stream>>>(zg, G * 64);
  k_pool<<<g1((long)N * 64), TPB, 0, stream>>>(act1, batch, zg, N, 64);

  // ================= head MLP =================
  k_mlp<<<1, 256, 0, stream>>>(hg, zg, fc1w, fc1b, fc2w, fc2b, fc3w, fc3b, (float*)d_out);
}
